// MultiHeadAttention_12807592477210
// MI455X (gfx1250) — matrix operand order repaired and checked by reading, not compiled, not run on hardware
//
#include <hip/hip_runtime.h>
#include <hip/hip_bf16.h>
#include <stdint.h>

typedef __bf16 bf16_t;
typedef __attribute__((ext_vector_type(16))) __bf16        v16bf;
typedef __attribute__((ext_vector_type(8)))  float         v8f;
typedef __attribute__((ext_vector_type(4)))  unsigned int  v4u;

static constexpr int kB   = 2;
static constexpr int kT   = 4096;
static constexpr int kC   = 768;
static constexpr int kH   = 12;
static constexpr int kD   = 64;
static constexpr int kM   = kB * kT;      // 8192 rows
static constexpr int kQB  = kT / 32;      // 128 q-blocks (32 rows) per head

union Frag {
    v16bf v;
    v4u   q[2];
};

__device__ __forceinline__ unsigned short f2bf_bits(float f) {
    unsigned u = __builtin_bit_cast(unsigned, f);
    unsigned r = (u + 0x7FFFu + ((u >> 16) & 1u)) >> 16;   // RNE
    return (unsigned short)r;
}
__device__ __forceinline__ bf16_t f2bf(float f) {
    unsigned short b = f2bf_bits(f);
    return __builtin_bit_cast(bf16_t, b);
}

__device__ __forceinline__ v8f wmma_bf16(const Frag& a, const Frag& b, v8f c) {
    return __builtin_amdgcn_wmma_f32_16x16x32_bf16(false, a.v, false, b.v,
                                                   (short)0, c, false, false);
}

// async copy 16B: LDS[lds_off] = MEM[g]; tracked by ASYNCcnt.
// VDST is a single 32-bit VGPR holding the wave-relative LDS byte offset
// (generic shared address truncated to 32 bits per the LDS aperture mapping).
__device__ __forceinline__ void async_b128(unsigned lds_off, const bf16_t* g) {
    asm volatile("global_load_async_to_lds_b128 %0, %1, off"
                 :: "v"(lds_off), "v"(g) : "memory");
}
__device__ __forceinline__ void wait_async0() {
    asm volatile("s_wait_asynccnt 0x0" ::: "memory");
}

// ---------------- prep: fp32 -> bf16 ----------------
__global__ void k_cvt(const float* __restrict__ in, bf16_t* __restrict__ out, int n) {
    for (int i = blockIdx.x * blockDim.x + threadIdx.x; i < n;
         i += gridDim.x * blockDim.x)
        out[i] = f2bf(in[i]);
}

// w[K][N] fp32 -> wt[N][K] bf16
__global__ void k_transpose_cvt(const float* __restrict__ w, bf16_t* __restrict__ wt,
                                int K, int N) {
    int total = K * N;
    for (int i = blockIdx.x * blockDim.x + threadIdx.x; i < total;
         i += gridDim.x * blockDim.x) {
        int nn = i / K, kk = i % K;
        wt[i] = f2bf(w[kk * N + nn]);
    }
}

// ---------------------------------------------------------------------------
// Block-cooperative GEMM core: 4 waves share one 64-col B panel staged in LDS
// via double-buffered async global->LDS copies. Each wave owns a 32x64 tile.
// Panel layout in LDS: [col][k] (32 k-values = 64B per col), matching the
// B-fragment layout (lane = col, half-wave h picks k 8h..8h+7 and 16+8h..16+8h+7).
// ---------------------------------------------------------------------------
__device__ __forceinline__ void gemm_core(const bf16_t* __restrict__ A,
                                          const bf16_t* __restrict__ Bt,
                                          unsigned short* bpanel,   // [2][64][32]
                                          int mt, int nb,
                                          int lane, int half, int n16,
                                          v8f acc[2][4]) {
    const int tid = threadIdx.x;               // 0..127
    // issue one 4KB panel (256 x 16B chunks, 2 per thread)
    auto issue = [&](int buf, int k0) {
#pragma unroll
        for (int c2 = 0; c2 < 2; ++c2) {
            const int c   = tid + c2 * 128;
            const int col = c >> 2, seg = c & 3;
            const unsigned lds_off =
                (unsigned)(size_t)(bpanel + buf * 2048 + col * 32 + seg * 8);
            async_b128(lds_off,
                       Bt + (size_t)(nb * 64 + col) * kC + k0 + seg * 8);
        }
    };

    const bf16_t* arow0 = A + (size_t)(mt * 32 + n16) * kC;
    const bf16_t* arow1 = arow0 + (size_t)16 * kC;

    issue(0, 0);
    for (int k0 = 0; k0 < kC; k0 += 32) {
        const int cur = (k0 >> 5) & 1;
        wait_async0();          // current panel landed in LDS
        __syncthreads();        // everyone ready; other buffer free for reuse
        if (k0 + 32 < kC) issue(cur ^ 1, k0 + 32);

        __builtin_prefetch(arow0 + k0 + 96, 0, 1);
        Frag fa[2];                                    // A 16x32: M=lane&15
        fa[0].q[0] = *(const v4u*)(arow0 + k0 + half * 8);
        fa[0].q[1] = *(const v4u*)(arow0 + k0 + 16 + half * 8);
        fa[1].q[0] = *(const v4u*)(arow1 + k0 + half * 8);
        fa[1].q[1] = *(const v4u*)(arow1 + k0 + 16 + half * 8);
#pragma unroll
        for (int j = 0; j < 4; ++j) {
            const unsigned short* bp =
                bpanel + cur * 2048 + (j * 16 + n16) * 32 + half * 8;
            Frag fb;                                   // B 32x16: N=lane&15
            fb.q[0] = *(const v4u*)(bp);
            fb.q[1] = *(const v4u*)(bp + 16);
            acc[0][j] = wmma_bf16(fa[0], fb, acc[0][j]);
            acc[1][j] = wmma_bf16(fa[1], fb, acc[1][j]);
        }
    }
}

// ---------------- GEMM1: qkv = x @ W + b, scattered to Q/K/V^T ----------------
__global__ void __launch_bounds__(128)
k_gemm_qkv(const bf16_t* __restrict__ A,    // x bf16 [8192,768]
           const bf16_t* __restrict__ Bt,   // qkv_w^T bf16 [2304,768]
           const float*  __restrict__ bias, // [2304]
           bf16_t* __restrict__ Qb,         // [B,H,T,D]
           bf16_t* __restrict__ Kb,         // [B,H,T,D]
           bf16_t* __restrict__ Vt) {       // [B,H,D,T]
    __shared__ unsigned short bpanel[2 * 64 * 32];     // 8KB double-buffered B panel
    const int lane = threadIdx.x & 31;
    const int half = lane >> 4;
    const int n16  = lane & 15;
    const int wIn  = threadIdx.x >> 5;
    const int NT   = (3 * kC) / 64;          // 36 n-blocks of 64
    const int nb   = blockIdx.x % NT;        // shared by all 4 waves
    const int mt   = (blockIdx.x / NT) * 4 + wIn;

    v8f acc[2][4];
    for (int i = 0; i < 2; ++i)
        for (int j = 0; j < 4; ++j)
            for (int e = 0; e < 8; ++e) acc[i][j][e] = 0.f;

    gemm_core(A, Bt, bpanel, mt, nb, lane, half, n16, acc);

#pragma unroll
    for (int j = 0; j < 4; ++j) {
        const int nn    = nb * 64 + j * 16 + n16;
        const float bv  = bias[nn];
        const int which = nn / kC;
        const int c     = nn % kC;
        const int h     = c / kD;
        const int d     = c % kD;
#pragma unroll
        for (int i = 0; i < 2; ++i) {
#pragma unroll
            for (int v = 0; v < 8; ++v) {              // C: M = v + 8*half
                const int r = mt * 32 + i * 16 + v + 8 * half;
                const int b = r / kT, t = r % kT;
                const int bh = b * kH + h;
                const bf16_t val = f2bf(acc[i][j][v] + bv);
                if (which == 0)      Qb[((size_t)bh * kT + t) * kD + d] = val;
                else if (which == 1) Kb[((size_t)bh * kT + t) * kD + d] = val;
                else                 Vt[((size_t)bh * kD + d) * kT + t] = val;
            }
        }
    }
}

// ---------------- flash attention: one wave per (b,h, 32-row q-block) ----------------
__global__ void __launch_bounds__(128)
k_attn(const bf16_t* __restrict__ Qb, const bf16_t* __restrict__ Kb,
       const bf16_t* __restrict__ Vt, bf16_t* __restrict__ Y) {
    __shared__ unsigned short sp[4 * 2 * 16 * 32];   // per wave: 2 P-tiles (16x32 bf16)
    const int lane = threadIdx.x & 31;
    const int half = lane >> 4;
    const int n16  = lane & 15;
    const int wIn  = threadIdx.x >> 5;
    unsigned short* lds = sp + wIn * (2 * 16 * 32);

    const int w    = blockIdx.x * 4 + wIn;
    const int qblk = w & (kQB - 1);          // 0..127
    const int bh   = w >> 7;                 // 0..23
    const bf16_t* Qp = Qb + (size_t)bh * kT * kD;
    const bf16_t* Kp = Kb + (size_t)bh * kT * kD;
    const bf16_t* Vp = Vt + (size_t)bh * kD * kT;

    Frag qa[2][2];                           // [row-tile i][d-half f]
#pragma unroll
    for (int i = 0; i < 2; ++i) {
        const bf16_t* qrow = Qp + (size_t)(qblk * 32 + i * 16 + n16) * kD;
        qa[i][0].q[0] = *(const v4u*)(qrow + half * 8);
        qa[i][0].q[1] = *(const v4u*)(qrow + 16 + half * 8);
        qa[i][1].q[0] = *(const v4u*)(qrow + 32 + half * 8);
        qa[i][1].q[1] = *(const v4u*)(qrow + 48 + half * 8);
    }

    v8f acc[2][4];
    float m_i[2][8], l_i[2][8];
#pragma unroll
    for (int i = 0; i < 2; ++i) {
        for (int j = 0; j < 4; ++j)
            for (int e = 0; e < 8; ++e) acc[i][j][e] = 0.f;
        for (int v = 0; v < 8; ++v) { m_i[i][v] = -1e30f; l_i[i][v] = 0.f; }
    }

    const int   kend  = (qblk + 1) * 32;
    const float scale = 0.125f;              // 1/sqrt(64)

    for (int k0 = 0; k0 < kend; k0 += 32) {
        __builtin_prefetch(Kp + (size_t)(k0 + 32 + n16) * kD, 0, 1);
        v8f s[2][2];                         // [row-tile i][key-subtile]
#pragma unroll
        for (int i = 0; i < 2; ++i)
            for (int sub = 0; sub < 2; ++sub)
                for (int e = 0; e < 8; ++e) s[i][sub][e] = 0.f;
#pragma unroll
        for (int sub = 0; sub < 2; ++sub) {  // K^T fragments shared by both row-tiles
            const int t = k0 + sub * 16 + n16;
#pragma unroll
            for (int f = 0; f < 2; ++f) {
                const bf16_t* kp = Kp + (size_t)t * kD + f * 32 + half * 8;
                Frag kf;
                kf.q[0] = *(const v4u*)(kp);
                kf.q[1] = *(const v4u*)(kp + 16);
                s[0][sub] = wmma_bf16(qa[0][f], kf, s[0][sub]);
                s[1][sub] = wmma_bf16(qa[1][f], kf, s[1][sub]);
            }
        }
        // online softmax per row-tile; exp values go straight to the LDS P-tile
#pragma unroll
        for (int i = 0; i < 2; ++i) {
            const int rel0 = k0 - (qblk * 32 + i * 16);
#pragma unroll
            for (int v = 0; v < 8; ++v) {
                const int Mrow = v + 8 * half;
                float s0 = s[i][0][v] * scale;
                float s1 = s[i][1][v] * scale;
                if (rel0 >= 0      && (n16 + rel0)      > Mrow) s0 = -1e30f;
                if (rel0 + 16 >= 0 && (n16 + rel0 + 16) > Mrow) s1 = -1e30f;
                float mx = fmaxf(s0, s1);
                for (int off = 1; off < 16; off <<= 1)
                    mx = fmaxf(mx, __shfl_xor(mx, off, 16));
                const float mn = fmaxf(m_i[i][v], mx);
                const float sc = __expf(m_i[i][v] - mn);
                const float a0 = __expf(s0 - mn);
                const float a1 = __expf(s1 - mn);
                float rs = a0 + a1;
                for (int off = 1; off < 16; off <<= 1)
                    rs += __shfl_xor(rs, off, 16);
                l_i[i][v] = l_i[i][v] * sc + rs;
                m_i[i][v] = mn;
#pragma unroll
                for (int j = 0; j < 4; ++j) acc[i][j][v] *= sc;
                const int rowOff = i * 512 + Mrow * 32;
                lds[rowOff + n16]      = f2bf_bits(a0);
                lds[rowOff + 16 + n16] = f2bf_bits(a1);
            }
        }
        // reload P in A-layout (same-wave LDS ops are in-order)
        Frag pa[2];
#pragma unroll
        for (int i = 0; i < 2; ++i) {
            pa[i].q[0] = *(const v4u*)(lds + i * 512 + n16 * 32 + half * 8);
            pa[i].q[1] = *(const v4u*)(lds + i * 512 + n16 * 32 + 16 + half * 8);
        }
        // Y += P @ V ; each V^T fragment feeds both row-tiles
#pragma unroll
        for (int j = 0; j < 4; ++j) {
            const bf16_t* vp = Vp + (size_t)(j * 16 + n16) * kT + k0 + half * 8;
            Frag vf;
            vf.q[0] = *(const v4u*)(vp);
            vf.q[1] = *(const v4u*)(vp + 16);
            acc[0][j] = wmma_bf16(pa[0], vf, acc[0][j]);
            acc[1][j] = wmma_bf16(pa[1], vf, acc[1][j]);
        }
    }

    const int b = bh / kH, h = bh % kH;
#pragma unroll
    for (int i = 0; i < 2; ++i) {
#pragma unroll
        for (int j = 0; j < 4; ++j) {
#pragma unroll
            for (int v = 0; v < 8; ++v) {
                const int t = qblk * 32 + i * 16 + v + 8 * half;
                const int c = h * kD + j * 16 + n16;
                Y[((size_t)(b * kT + t)) * kC + c] = f2bf(acc[i][j][v] / l_i[i][v]);
            }
        }
    }
}

// ---------------- GEMM2: out = y @ out_w + out_b (fp32 out) ----------------
__global__ void __launch_bounds__(128)
k_gemm_out(const bf16_t* __restrict__ A,    // y bf16 [8192,768]
           const bf16_t* __restrict__ Bt,   // out_w^T bf16 [768,768]
           const float*  __restrict__ bias, // [768]
           float* __restrict__ Out) {
    __shared__ unsigned short bpanel[2 * 64 * 32];     // 8KB double-buffered B panel
    const int lane = threadIdx.x & 31;
    const int half = lane >> 4;
    const int n16  = lane & 15;
    const int wIn  = threadIdx.x >> 5;
    const int NT   = kC / 64;                // 12
    const int nb   = blockIdx.x % NT;
    const int mt   = (blockIdx.x / NT) * 4 + wIn;

    v8f acc[2][4];
    for (int i = 0; i < 2; ++i)
        for (int j = 0; j < 4; ++j)
            for (int e = 0; e < 8; ++e) acc[i][j][e] = 0.f;

    gemm_core(A, Bt, bpanel, mt, nb, lane, half, n16, acc);

#pragma unroll
    for (int j = 0; j < 4; ++j) {
        const int nn   = nb * 64 + j * 16 + n16;
        const float bv = bias[nn];
#pragma unroll
        for (int i = 0; i < 2; ++i) {
#pragma unroll
            for (int v = 0; v < 8; ++v) {
                const int r = mt * 32 + i * 16 + v + 8 * half;
                Out[(size_t)r * kC + nn] = acc[i][j][v] + bv;
            }
        }
    }
}

extern "C" void kernel_launch(void* const* d_in, const int* in_sizes, int n_in,
                              void* d_out, int out_size, void* d_ws, size_t ws_size,
                              hipStream_t stream) {
    const float* x     = (const float*)d_in[0];
    const float* qkv_w = (const float*)d_in[1];
    const float* qkv_b = (const float*)d_in[2];
    const float* out_w = (const float*)d_in[3];
    const float* out_b = (const float*)d_in[4];
    float* out = (float*)d_out;

    char*  ws  = (char*)d_ws;
    size_t off = 0;
    auto carve = [&](size_t bytes) -> char* {
        char* p = ws + off;
        off += (bytes + 255) & ~(size_t)255;
        return p;
    };
    bf16_t* xbf = (bf16_t*)carve((size_t)kM * kC * 2);       // x bf16 (reused as y)
    bf16_t* wt1 = (bf16_t*)carve((size_t)3 * kC * kC * 2);   // qkv_w^T
    bf16_t* wt2 = (bf16_t*)carve((size_t)kC * kC * 2);       // out_w^T
    bf16_t* qb  = (bf16_t*)carve((size_t)kM * kC * 2);       // [B,H,T,D]
    bf16_t* kbuf= (bf16_t*)carve((size_t)kM * kC * 2);       // [B,H,T,D]
    bf16_t* vt  = (bf16_t*)carve((size_t)kM * kC * 2);       // [B,H,D,T]
    bf16_t* y   = xbf;  // x is dead after GEMM1; stream order makes this safe

    k_cvt<<<2048, 256, 0, stream>>>(x, xbf, kM * kC);
    k_transpose_cvt<<<2048, 256, 0, stream>>>(qkv_w, wt1, kC, 3 * kC);
    k_transpose_cvt<<<1024, 256, 0, stream>>>(out_w, wt2, kC, kC);

    // 64 m-groups (128 rows) x 36 n-blocks = 2304 blocks, 4 waves/block share B panel
    k_gemm_qkv<<<2304, 128, 0, stream>>>(xbf, wt1, qkv_b, qb, kbuf, vt);

    // 24 heads x 128 q-blocks (32 rows) = 3072 waves
    k_attn<<<768, 128, 0, stream>>>(qb, kbuf, vt, y);

    // 64 m-groups x 12 n-blocks = 768 blocks
    k_gemm_out<<<768, 128, 0, stream>>>(y, wt2, out_b, out);
}
